// QwenVL_PartB_20968030339728
// MI455X (gfx1250) — matrix operand order repaired and checked by reading, not compiled, not run on hardware
//
#include <hip/hip_runtime.h>
#include <hip/hip_bf16.h>

// ---------------------------------------------------------------------------
// Qwen-VL vision tower (part B) on gfx1250 / MI455X.
// All GEMMs via v_wmma_f32_16x16x32_f16 (wave32). fp32 weights converted to
// f16 on the fly; activations kept f16 (padded) in workspace.
// Wave tile 32x64 (2x4 fragments, 8 WMMA/K-step); block = 8 waves = 64x256.
// K-loop split into branch-free main body + guarded (select-based) tail.
// ---------------------------------------------------------------------------

typedef __attribute__((ext_vector_type(16))) _Float16 v16h;
typedef __attribute__((ext_vector_type(8)))  _Float16 v8h;
typedef __attribute__((ext_vector_type(8)))  float    v8f;
typedef __attribute__((ext_vector_type(4)))  float    v4f;

#define SEQ      900      // real tokens
#define SPAD     912      // tokens padded to 16
#define SP28     928      // tokens padded to 32 (K-dim use)
#define NHEAD    16
#define HDIM     80
#define HDP      96       // head dim padded to 32
#define DMODEL   1280
#define FF       3420
#define FFP      3424
#define KPATCH   2352
#define KPATCHP  2368
#define MERGED   5120
#define MROWS    240
#define MREAL    225
#define OUTD     3584
#define NLAYER   8
#define HCH      4        // heads per attention chunk
#define QKSCALE  0.3343701525f   // 80^-0.25
#define EPSF     1e-6f

// ---------------------------------------------------------------------------
// Generic WMMA GEMM:  C[m,n] (+)= sum_k A[m,k] * B[n,k] (+ bias[n])
//   A: f16, lda (padded K, multiple of 32), rows clamped to M_ld for loads
//   B: f16 (padded, ldb mult of 32) or f32 (exact K, tail-guarded)
//   EPI: 0 = C f32 = acc+bias ; 1 = C f32 += acc+bias ; 2 = C f16 = acc
// ---------------------------------------------------------------------------
template<bool B_IS_F16, int EPI>
__global__ __launch_bounds__(256) void gemm_wmma(
    const _Float16* __restrict__ A, const void* __restrict__ Bv,
    const float* __restrict__ bias, void* __restrict__ Cv,
    int M_ld, int Mst, int N_ld, int Nst, int Kreal, int KP,
    int lda, int ldb, int ldc,
    long long sA, long long sB, long long sC)
{
  const int tid  = threadIdx.x;
  const int wid  = tid >> 5;
  const int lane = tid & 31;
  const int l16  = lane & 15;
  const int hi   = lane >> 4;
  const int wm   = wid >> 2;       // 0..1
  const int wn   = wid & 3;        // 0..3
  const int z    = blockIdx.z;

  const _Float16* Ab  = A + (size_t)z * sA;
  const _Float16* B16 = (const _Float16*)Bv + (size_t)z * sB;
  const float*    B32 = (const float*)Bv    + (size_t)z * sB;
  const size_t    cz  = (size_t)z * sC;

  const int m0 = blockIdx.y * 64  + wm * 32;
  const int n0 = blockIdx.x * 256 + wn * 64;

  v8f c[2][4];
#pragma unroll
  for (int i = 0; i < 2; ++i)
#pragma unroll
    for (int j = 0; j < 4; ++j)
#pragma unroll
      for (int e = 0; e < 8; ++e) c[i][j][e] = 0.0f;

  int ra[2], rb[4];
#pragma unroll
  for (int i = 0; i < 2; ++i) {
    int r = m0 + 16 * i + l16;
    ra[i] = (r < M_ld) ? r : (M_ld - 1);
  }
#pragma unroll
  for (int j = 0; j < 4; ++j) {
    int r = n0 + 16 * j + l16;
    rb[j] = (r < N_ld) ? r : (N_ld - 1);
  }

  const int kfast = Kreal & ~31;   // main-loop bound: all 16-wide B reads valid
  int kb = 0;

  // ====================== branch-free main K loop ==========================
  for (; kb < kfast; kb += 32) {
    v16h a[2], b[4];
#pragma unroll
    for (int i = 0; i < 2; ++i) {
      const _Float16* p = Ab + (size_t)ra[i] * lda + kb + 16 * hi;
      v8h lo  = *(const v8h*)p;          // K = kb+16hi .. +7    -> VGPR 0..3
      v8h hi8 = *(const v8h*)(p + 8);    // K = kb+16hi+8 .. +7 -> VGPR 4..7
#pragma unroll
      for (int e = 0; e < 8; ++e) { a[i][e] = lo[e]; a[i][8 + e] = hi8[e]; }
    }
    const int ks = kb + 16 * hi;
    if constexpr (B_IS_F16) {
#pragma unroll
      for (int j = 0; j < 4; ++j)
        b[j] = *(const v16h*)(B16 + (size_t)rb[j] * ldb + ks);
    } else {
#pragma unroll
      for (int j = 0; j < 4; ++j) {
        const float* p = B32 + (size_t)rb[j] * ldb + ks;
        v4f f0 = *(const v4f*)(p);
        v4f f1 = *(const v4f*)(p + 4);
        v4f f2 = *(const v4f*)(p + 8);
        v4f f3 = *(const v4f*)(p + 12);
#pragma unroll
        for (int e = 0; e < 4; ++e) {
          b[j][e]      = (_Float16)f0[e];
          b[j][4 + e]  = (_Float16)f1[e];
          b[j][8 + e]  = (_Float16)f2[e];
          b[j][12 + e] = (_Float16)f3[e];
        }
      }
    }
    if (kb + 32 < kfast) {
      __builtin_prefetch((const void*)(Ab + (size_t)ra[0] * lda + kb + 32), 0, 1);
      if constexpr (B_IS_F16)
        __builtin_prefetch((const void*)(B16 + (size_t)rb[0] * ldb + kb + 32), 0, 1);
      else
        __builtin_prefetch((const void*)(B32 + (size_t)rb[0] * ldb + kb + 32), 0, 1);
    }
#pragma unroll
    for (int i = 0; i < 2; ++i)
#pragma unroll
      for (int j = 0; j < 4; ++j)
        c[i][j] = __builtin_amdgcn_wmma_f32_16x16x32_f16(
            false, a[i], false, b[j], (short)0, c[i][j], false, false);
  }

  // ====================== guarded tail (select-based, no exec branch) ======
  for (; kb < KP; kb += 32) {
    v16h a[2], b[4];
#pragma unroll
    for (int i = 0; i < 2; ++i) {
      const _Float16* p = Ab + (size_t)ra[i] * lda + kb + 16 * hi;
      v8h lo  = *(const v8h*)p;
      v8h hi8 = *(const v8h*)(p + 8);
#pragma unroll
      for (int e = 0; e < 8; ++e) { a[i][e] = lo[e]; a[i][8 + e] = hi8[e]; }
    }
    const int ks = kb + 16 * hi;
    if constexpr (B_IS_F16) {
#pragma unroll
      for (int j = 0; j < 4; ++j)
        b[j] = *(const v16h*)(B16 + (size_t)rb[j] * ldb + ks);
    } else {
#pragma unroll
      for (int j = 0; j < 4; ++j) {
        const float* p = B32 + (size_t)rb[j] * ldb + ks;
#pragma unroll
        for (int e = 0; e < 16; ++e) {
          int k = ks + e;
          b[j][e] = (k < Kreal) ? (_Float16)p[e] : (_Float16)0.0f;
        }
      }
    }
#pragma unroll
    for (int i = 0; i < 2; ++i)
#pragma unroll
      for (int j = 0; j < 4; ++j)
        c[i][j] = __builtin_amdgcn_wmma_f32_16x16x32_f16(
            false, a[i], false, b[j], (short)0, c[i][j], false, false);
  }

  // ---- epilogue: C layout lane=(N col), VGPR r -> M = r + 8*hi ------------
#pragma unroll
  for (int i = 0; i < 2; ++i) {
#pragma unroll
    for (int j = 0; j < 4; ++j) {
      int col = n0 + 16 * j + l16;
      if (col >= Nst) continue;
      float bv = bias ? bias[col] : 0.0f;
#pragma unroll
      for (int r = 0; r < 8; ++r) {
        int row = m0 + 16 * i + 8 * hi + r;
        if (row >= Mst) continue;
        float v = c[i][j][r] + bv;
        size_t idx = cz + (size_t)row * ldc + col;
        if constexpr (EPI == 0)      ((float*)Cv)[idx] = v;
        else if constexpr (EPI == 1) ((float*)Cv)[idx] += v;
        else                         ((_Float16*)Cv)[idx] = (_Float16)v;
      }
    }
  }
}

// ---------------------------------------------------------------------------
// Preprocess: bilinear 512->420, normalize, patchify -> X16 [SPAD x KPATCHP]
// ---------------------------------------------------------------------------
__global__ __launch_bounds__(256) void k_preprocess(
    const float* __restrict__ px, _Float16* __restrict__ X)
{
  int idx = blockIdx.x * 256 + threadIdx.x;
  if (idx >= SPAD * KPATCHP) return;
  int f = idx % KPATCHP, s = idx / KPATCHP;
  float v = 0.0f;
  if (s < SEQ && f < KPATCH) {
    int p2 = f % 14, p1 = (f / 14) % 14, ch = f / 392;   // t-index duplicate
    int ms2 = s & 1, ms1 = (s >> 1) & 1, wf = (s >> 2) % 15, hf = s / 60;
    int row = (hf * 2 + ms1) * 14 + p1;
    int col = (wf * 2 + ms2) * 14 + p2;
    float fy = row * (511.0f / 419.0f), fx = col * (511.0f / 419.0f);
    int y0 = (int)fy, x0 = (int)fx;
    int y1 = (y0 + 1 < 512) ? y0 + 1 : 511;
    int x1 = (x0 + 1 < 512) ? x0 + 1 : 511;
    float wy = fy - (float)y0, wx = fx - (float)x0;
    const float* cp = px + (size_t)ch * 512 * 512;
    float a = cp[y0 * 512 + x0], b = cp[y0 * 512 + x1];
    float cc = cp[y1 * 512 + x0], dd = cp[y1 * 512 + x1];
    float bil = (a * (1.f - wx) + b * wx) * (1.f - wy)
              + (cc * (1.f - wx) + dd * wx) * wy;
    float inv  = (ch == 0) ? (1.f / 0.26862954f)
               : (ch == 1) ? (1.f / 0.26130258f) : (1.f / 0.27577711f);
    float mean = (ch == 0) ? 0.48145466f : (ch == 1) ? 0.4578275f : 0.40821073f;
    v = bil * (inv / 255.0f) - mean * inv;
  }
  X[idx] = (_Float16)v;
}

// ---------------------------------------------------------------------------
// RMSNorm -> f16 (pad rows zeroed). grid = SPAD rows.
// ---------------------------------------------------------------------------
__global__ __launch_bounds__(256) void k_rmsnorm(
    const float* __restrict__ H, const float* __restrict__ w,
    _Float16* __restrict__ out)
{
  __shared__ float red[256];
  int s = blockIdx.x;
  const float* row = H + (size_t)s * DMODEL;
  float ss = 0.0f;
  for (int d = threadIdx.x; d < DMODEL; d += 256) { float v = row[d]; ss += v * v; }
  red[threadIdx.x] = ss; __syncthreads();
  for (int off = 128; off > 0; off >>= 1) {
    if (threadIdx.x < off) red[threadIdx.x] += red[threadIdx.x + off];
    __syncthreads();
  }
  float rinv = rsqrtf(red[0] / (float)DMODEL + EPSF);
  bool live = (s < SEQ);
  for (int d = threadIdx.x; d < DMODEL; d += 256)
    out[(size_t)s * DMODEL + d] = live ? (_Float16)(w[d] * row[d] * rinv)
                                       : (_Float16)0.0f;
}

// Final RMS + 2x2 merge reshape -> M16 [MROWS x MERGED]. grid = 960 rows.
__global__ __launch_bounds__(256) void k_merge_rms(
    const float* __restrict__ H, const float* __restrict__ w,
    _Float16* __restrict__ M16)
{
  __shared__ float red[256];
  int s = blockIdx.x;                       // 0..959
  int sl = (s < SPAD) ? s : (SPAD - 1);
  const float* row = H + (size_t)sl * DMODEL;
  float ss = 0.0f;
  for (int d = threadIdx.x; d < DMODEL; d += 256) { float v = row[d]; ss += v * v; }
  red[threadIdx.x] = ss; __syncthreads();
  for (int off = 128; off > 0; off >>= 1) {
    if (threadIdx.x < off) red[threadIdx.x] += red[threadIdx.x + off];
    __syncthreads();
  }
  float rinv = rsqrtf(red[0] / (float)DMODEL + EPSF);
  bool live = (s < SEQ);
  size_t base = (size_t)(s >> 2) * MERGED + (size_t)(s & 3) * DMODEL;
  for (int d = threadIdx.x; d < DMODEL; d += 256)
    M16[base + d] = live ? (_Float16)(w[d] * row[d] * rinv) : (_Float16)0.0f;
}

// ---------------------------------------------------------------------------
// QKV post: scale + RoPE -> Q16/K16 [h][SPAD][HDP]; V transpose -> VT16
// [h][HDIM][SP28]. grid covers NHEAD*SP28*HDP threads.
// ---------------------------------------------------------------------------
__global__ __launch_bounds__(256) void k_qkvpost(
    const float* __restrict__ QKV, const float* __restrict__ cosT,
    const float* __restrict__ sinT, _Float16* __restrict__ Q16,
    _Float16* __restrict__ K16, _Float16* __restrict__ VT16)
{
  int idx = blockIdx.x * 256 + threadIdx.x;
  if (idx >= NHEAD * SP28 * HDP) return;
  int d = idx % HDP;
  int s = (idx / HDP) % SP28;
  int h = idx / (HDP * SP28);

  if (s < SPAD) {
    _Float16 qv = (_Float16)0.0f, kv = (_Float16)0.0f;
    if (d < HDIM) {
      const float* row = QKV + (size_t)s * (3 * DMODEL);
      int sc = (s < SEQ) ? s : (SEQ - 1);
      float cv = cosT[sc * HDIM + d], sv = sinT[sc * HDIM + d];
      int dr = (d < 40) ? d + 40 : d - 40;
      float sgn = (d < 40) ? -1.0f : 1.0f;
      float q  = row[h * HDIM + d]  * QKSCALE;
      float qr = row[h * HDIM + dr] * QKSCALE * sgn;
      float k  = row[DMODEL + h * HDIM + d]  * QKSCALE;
      float kr = row[DMODEL + h * HDIM + dr] * QKSCALE * sgn;
      qv = (_Float16)(q * cv + qr * sv);
      kv = (_Float16)(k * cv + kr * sv);
    }
    Q16[((size_t)h * SPAD + s) * HDP + d] = qv;
    K16[((size_t)h * SPAD + s) * HDP + d] = kv;
  }
  if (d < HDIM) {
    float v = 0.0f;
    if (s < SPAD) v = QKV[(size_t)s * (3 * DMODEL) + 2 * DMODEL + h * HDIM + d];
    VT16[((size_t)h * HDIM + d) * SP28 + s] = (_Float16)v;
  }
}

// ---------------------------------------------------------------------------
// Softmax over t (with mask add) -> f16 attn rows (pad zeroed).
// grid = (SPAD, HCH)
// ---------------------------------------------------------------------------
__global__ __launch_bounds__(256) void k_softmax(
    const float* __restrict__ SC, const float* __restrict__ mask,
    _Float16* __restrict__ AT)
{
  int s = blockIdx.x, zh = blockIdx.y;
  const float* srow = SC + ((size_t)zh * SPAD + s) * SP28;
  _Float16*    orow = AT + ((size_t)zh * SPAD + s) * SP28;
  if (s >= SEQ) {
    for (int t = threadIdx.x; t < SP28; t += 256) orow[t] = (_Float16)0.0f;
    return;
  }
  const float* mrow = mask + (size_t)s * SEQ;
  __shared__ float red[256];
  float mx = -3.0e38f;
  for (int t = threadIdx.x; t < SEQ; t += 256)
    mx = fmaxf(mx, srow[t] + mrow[t]);
  red[threadIdx.x] = mx; __syncthreads();
  for (int off = 128; off > 0; off >>= 1) {
    if (threadIdx.x < off)
      red[threadIdx.x] = fmaxf(red[threadIdx.x], red[threadIdx.x + off]);
    __syncthreads();
  }
  mx = red[0]; __syncthreads();
  float sum = 0.0f;
  for (int t = threadIdx.x; t < SEQ; t += 256)
    sum += __expf(srow[t] + mrow[t] - mx);
  red[threadIdx.x] = sum; __syncthreads();
  for (int off = 128; off > 0; off >>= 1) {
    if (threadIdx.x < off) red[threadIdx.x] += red[threadIdx.x + off];
    __syncthreads();
  }
  float inv = 1.0f / red[0];
  for (int t = threadIdx.x; t < SP28; t += 256)
    orow[t] = (t < SEQ) ? (_Float16)(__expf(srow[t] + mrow[t] - mx) * inv)
                        : (_Float16)0.0f;
}

// SwiGLU: GU16 = f16( silu(G) * U ), padded cols zeroed.
__global__ __launch_bounds__(256) void k_swiglu(
    const float* __restrict__ G, const float* __restrict__ U,
    _Float16* __restrict__ O)
{
  int idx = blockIdx.x * 256 + threadIdx.x;
  if (idx >= SPAD * FFP) return;
  int f = idx % FFP, s = idx / FFP;
  float v = 0.0f;
  if (f < FF) {
    float g = G[(size_t)s * FF + f];
    float u = U[(size_t)s * FF + f];
    v = g / (1.0f + __expf(-g)) * u;
  }
  O[idx] = (_Float16)v;
}

// Exact GELU -> f16
__global__ __launch_bounds__(256) void k_gelu(
    const float* __restrict__ T, _Float16* __restrict__ O)
{
  int idx = blockIdx.x * 256 + threadIdx.x;
  if (idx >= MROWS * MERGED) return;
  float x = T[idx];
  O[idx] = (_Float16)(0.5f * x * (1.0f + erff(x * 0.70710678f)));
}

// ---------------------------------------------------------------------------
extern "C" void kernel_launch(void* const* d_in, const int* in_sizes, int n_in,
                              void* d_out, int out_size, void* d_ws, size_t ws_size,
                              hipStream_t stream)
{
  (void)in_sizes; (void)n_in; (void)out_size; (void)ws_size;
  const float* px      = (const float*)d_in[0];
  const float* patch_w = (const float*)d_in[1];
  const float* norm1_w = (const float*)d_in[2];
  const float* qkv_w   = (const float*)d_in[3];
  const float* qkv_b   = (const float*)d_in[4];
  const float* proj_w  = (const float*)d_in[5];
  const float* proj_b  = (const float*)d_in[6];
  const float* norm2_w = (const float*)d_in[7];
  const float* gate_w  = (const float*)d_in[8];
  const float* gate_b  = (const float*)d_in[9];
  const float* up_w    = (const float*)d_in[10];
  const float* up_b    = (const float*)d_in[11];
  const float* down_w  = (const float*)d_in[12];
  const float* down_b  = (const float*)d_in[13];
  const float* lnq_w   = (const float*)d_in[14];
  const float* m1_w    = (const float*)d_in[15];
  const float* m1_b    = (const float*)d_in[16];
  const float* m2_w    = (const float*)d_in[17];
  const float* m2_b    = (const float*)d_in[18];
  const float* cosT    = (const float*)d_in[19];
  const float* sinT    = (const float*)d_in[20];
  const float* masksP  = (const float*)d_in[21];

  // ---- workspace layout (≈97 MB) ------------------------------------------
  size_t off = 0;
  auto wsa = [&](size_t elems, size_t esz) -> void* {
    void* p = (char*)d_ws + off;
    off += (elems * esz + 255) & ~(size_t)255;
    return p;
  };
  _Float16* X16  = (_Float16*)wsa((size_t)SPAD * KPATCHP, 2);
  float*    H    = (float*)   wsa((size_t)SPAD * DMODEL, 4);
  _Float16* HN16 = (_Float16*)wsa((size_t)SPAD * DMODEL, 2);
  float*    QKV  = (float*)   wsa((size_t)SPAD * 3 * DMODEL, 4);
  _Float16* Q16  = (_Float16*)wsa((size_t)NHEAD * SPAD * HDP, 2);
  _Float16* K16  = (_Float16*)wsa((size_t)NHEAD * SPAD * HDP, 2);
  _Float16* VT16 = (_Float16*)wsa((size_t)NHEAD * HDIM * SP28, 2);
  float*    SC   = (float*)   wsa((size_t)HCH * SPAD * SP28, 4);
  _Float16* AT16 = (_Float16*)wsa((size_t)HCH * SPAD * SP28, 2);
  _Float16* O16  = (_Float16*)wsa((size_t)SPAD * DMODEL, 2);
  float*    G    = (float*)   wsa((size_t)SPAD * FF, 4);
  float*    U    = (float*)   wsa((size_t)SPAD * FF, 4);
  _Float16* GU16 = (_Float16*)wsa((size_t)SPAD * FFP, 2);
  _Float16* M16  = (_Float16*)wsa((size_t)MROWS * MERGED, 2);
  float*    T1   = (float*)   wsa((size_t)MROWS * MERGED, 4);
  _Float16* T116 = (_Float16*)wsa((size_t)MROWS * MERGED, 2);

  const int MT = (SPAD + 63) / 64;   // 15 M-tiles for token GEMMs
  #define NT(n) (((n) + 255) / 256)  // 256-wide N tiles

  // ---- patchify + patch GEMM ---------------------------------------------
  k_preprocess<<<(SPAD * KPATCHP + 255) / 256, 256, 0, stream>>>(px, X16);
  gemm_wmma<false, 0><<<dim3(NT(DMODEL), MT, 1), 256, 0, stream>>>(
      X16, patch_w, nullptr, H, SPAD, SPAD, DMODEL, DMODEL,
      KPATCH, KPATCHP, KPATCHP, KPATCH, DMODEL, 0, 0, 0);

  // ---- transformer blocks -------------------------------------------------
  for (int l = 0; l < NLAYER; ++l) {
    const float* qw = qkv_w  + (size_t)l * 3 * DMODEL * DMODEL;
    const float* qb = qkv_b  + (size_t)l * 3 * DMODEL;
    const float* pw = proj_w + (size_t)l * DMODEL * DMODEL;
    const float* pb = proj_b + (size_t)l * DMODEL;
    const float* gw = gate_w + (size_t)l * FF * DMODEL;
    const float* gb = gate_b + (size_t)l * FF;
    const float* uw = up_w   + (size_t)l * FF * DMODEL;
    const float* ub = up_b   + (size_t)l * FF;
    const float* dw = down_w + (size_t)l * DMODEL * FF;
    const float* db = down_b + (size_t)l * DMODEL;
    const float* mk = masksP + (size_t)l * SEQ * SEQ;

    k_rmsnorm<<<SPAD, 256, 0, stream>>>(H, norm1_w + (size_t)l * DMODEL, HN16);
    gemm_wmma<false, 0><<<dim3(NT(3 * DMODEL), MT, 1), 256, 0, stream>>>(
        HN16, qw, qb, QKV, SPAD, SPAD, 3 * DMODEL, 3 * DMODEL,
        DMODEL, DMODEL, DMODEL, DMODEL, 3 * DMODEL, 0, 0, 0);
    k_qkvpost<<<(NHEAD * SP28 * HDP + 255) / 256, 256, 0, stream>>>(
        QKV, cosT, sinT, Q16, K16, VT16);

    for (int ch = 0; ch < NHEAD / HCH; ++ch) {
      const _Float16* Qc = Q16  + (size_t)ch * HCH * SPAD * HDP;
      const _Float16* Kc = K16  + (size_t)ch * HCH * SPAD * HDP;
      const _Float16* Vc = VT16 + (size_t)ch * HCH * HDIM * SP28;
      // scores = Q @ K^T (per head)
      gemm_wmma<true, 0><<<dim3(NT(SPAD), MT, HCH), 256, 0, stream>>>(
          Qc, Kc, nullptr, SC, SPAD, SPAD, SPAD, SPAD,
          HDP, HDP, HDP, HDP, SP28,
          (long long)SPAD * HDP, (long long)SPAD * HDP, (long long)SPAD * SP28);
      k_softmax<<<dim3(SPAD, HCH), 256, 0, stream>>>(SC, mk, AT16);
      // O = attn @ V  (f16 out, per-head column offset in O16)
      gemm_wmma<true, 2><<<dim3(1, MT, HCH), 256, 0, stream>>>(
          AT16, Vc, nullptr, O16 + (size_t)ch * HCH * HDIM,
          SPAD, SPAD, HDIM, HDIM, SP28, SP28, SP28, SP28, DMODEL,
          (long long)SPAD * SP28, (long long)HDIM * SP28, (long long)HDIM);
    }

    // H += O @ proj^T + pb
    gemm_wmma<false, 1><<<dim3(NT(DMODEL), MT, 1), 256, 0, stream>>>(
        O16, pw, pb, H, SPAD, SPAD, DMODEL, DMODEL,
        DMODEL, DMODEL, DMODEL, DMODEL, DMODEL, 0, 0, 0);

    // MLP
    k_rmsnorm<<<SPAD, 256, 0, stream>>>(H, norm2_w + (size_t)l * DMODEL, HN16);
    gemm_wmma<false, 0><<<dim3(NT(FF), MT, 1), 256, 0, stream>>>(
        HN16, gw, gb, G, SPAD, SPAD, FF, FF,
        DMODEL, DMODEL, DMODEL, DMODEL, FF, 0, 0, 0);
    gemm_wmma<false, 0><<<dim3(NT(FF), MT, 1), 256, 0, stream>>>(
        HN16, uw, ub, U, SPAD, SPAD, FF, FF,
        DMODEL, DMODEL, DMODEL, DMODEL, FF, 0, 0, 0);
    k_swiglu<<<(SPAD * FFP + 255) / 256, 256, 0, stream>>>(G, U, GU16);
    gemm_wmma<false, 1><<<dim3(NT(DMODEL), MT, 1), 256, 0, stream>>>(
        GU16, dw, db, H, SPAD, SPAD, DMODEL, DMODEL,
        FF, FFP, FFP, FF, DMODEL, 0, 0, 0);
  }

  // ---- merger -------------------------------------------------------------
  k_merge_rms<<<960, 256, 0, stream>>>(H, lnq_w, M16);
  gemm_wmma<false, 0><<<dim3(NT(MERGED), (MROWS + 63) / 64, 1), 256, 0, stream>>>(
      M16, m1_w, m1_b, T1, MROWS, MROWS, MERGED, MERGED,
      MERGED, MERGED, MERGED, MERGED, MERGED, 0, 0, 0);
  k_gelu<<<(MROWS * MERGED + 255) / 256, 256, 0, stream>>>(T1, T116);
  gemm_wmma<false, 0><<<dim3(NT(OUTD), (MROWS + 63) / 64, 1), 256, 0, stream>>>(
      T116, m2_w, m2_b, (float*)d_out, MROWS, MREAL, OUTD, OUTD,
      MERGED, MERGED, MERGED, MERGED, OUTD, 0, 0, 0);
  #undef NT
}
